// TransDecoder2_19344532702252
// MI455X (gfx1250) — matrix operand order repaired and checked by reading, not compiled, not run on hardware
//
#include <hip/hip_runtime.h>
#include <hip/hip_bf16.h>

// ---------------------------------------------------------------------------
// TransDecoder2: Q=MLP(x), K=MLP(c), V=MLP(c); out = sigmoid(Q K^T) V
// f16 WMMA (v_wmma_f32_16x16x32_f16) for all 512-wide GEMMs and for Q K^T.
// Round 3: register blocking resized to fit the 256 addressable VGPRs
// (round 2 spilled to scratch). GEMM = 32x64 tile/wave (2x4 subtiles,
// ~150 VGPRs peak). Attention = 16 Q rows/wave with the full 16x512 Q tile
// hoisted to registers (128 VGPRs), K streamed; sigmoid + A@V fused.
// ---------------------------------------------------------------------------

typedef _Float16 half_t;
typedef __attribute__((ext_vector_type(16))) _Float16 v16h;
typedef __attribute__((ext_vector_type(8)))  _Float16 v8h;
typedef __attribute__((ext_vector_type(8)))  float    v8f;

#define NX_   8192
#define NC_   8192
#define HD_   512
#define KD_   512
#define YD_   3
#define NHID_ 3        // N_LAYERS - 1 hidden 512x512 layers
#define NSPLIT_ 16     // column splits in fused attention

// Build a 16-half fragment from two contiguous 8-half (16B) pieces.
static __device__ __forceinline__ v16h load16(const half_t* p0, const half_t* p1) {
  v8h lo = *(const v8h*)p0;
  v8h hi = *(const v8h*)p1;
  v16h r;
#pragma unroll
  for (int i = 0; i < 8; ++i) { r[i] = lo[i]; r[i + 8] = hi[i]; }
  return r;
}

// --------------------------- utility kernels -------------------------------

__global__ void k_zero_f32(float* __restrict__ p, int n) {
  int t = blockIdx.x * blockDim.x + threadIdx.x;
  if (t < n) p[t] = 0.0f;
}

// f32 [in][out] -> f16 transposed [out][in] (so B-fragments are contiguous)
__global__ void k_cvt_t(const float* __restrict__ W, half_t* __restrict__ Wt,
                        int in_dim, int out_dim) {
  int t = blockIdx.x * blockDim.x + threadIdx.x;
  if (t >= in_dim * out_dim) return;
  int i = t / out_dim, o = t % out_dim;
  Wt[(size_t)o * in_dim + i] = (half_t)W[t];
}

// Layer 0: K-dim is only 3 -> plain VALU. Y = sin(X @ W0 + b0), f16 out.
__global__ void k_layer0(const float* __restrict__ X, const float* __restrict__ W0,
                         const float* __restrict__ b0, half_t* __restrict__ Y, int N) {
  int t = blockIdx.x * blockDim.x + threadIdx.x;
  if (t >= N * HD_) return;
  int nrow = t / HD_, j = t % HD_;
  float val = b0[j]
            + X[nrow * 3 + 0] * W0[0 * HD_ + j]
            + X[nrow * 3 + 1] * W0[1 * HD_ + j]
            + X[nrow * 3 + 2] * W0[2 * HD_ + j];
  Y[(size_t)nrow * HD_ + j] = (half_t)__sinf(val);
}

// V output head: 512 -> 3, plain VALU dot products, f32 out.
__global__ void k_vout(const half_t* __restrict__ H, const float* __restrict__ Wo,
                       const float* __restrict__ bo, float* __restrict__ V, int N) {
  int t = blockIdx.x * blockDim.x + threadIdx.x;
  if (t >= N * YD_) return;
  int nrow = t / YD_, y = t % YD_;
  float sum = bo[y];
#pragma unroll 8
  for (int k2 = 0; k2 < HD_; ++k2)
    sum += (float)H[(size_t)nrow * HD_ + k2] * Wo[k2 * YD_ + y];
  V[t] = sum;
}

// ----------------------- WMMA GEMM (+bias, +sin) ---------------------------
// Y[N,Nout] = act(X[N,Kdim] @ W[Kdim,Nout] + bias), W passed f16-transposed
// as Wt[Nout][Kdim]. One wave computes a 32x64 output tile = 2x4 WMMA
// subtiles: per K-step, 2 A-frags + 4 B-frags feed 8 WMMAs. ~150 VGPRs peak
// -> no scratch spills, still 2.67x the fetch efficiency of 1x1 tiling.
__global__ void __launch_bounds__(256, 1)
k_gemm_sin(const half_t* __restrict__ X, const half_t* __restrict__ Wt,
           const float* __restrict__ bias, half_t* __restrict__ Y,
           int N, int Kdim, int Nout, int applySin) {
  const int wave = blockIdx.x * (blockDim.x >> 5) + (threadIdx.x >> 5);
  const int lane = threadIdx.x & 31;
  const int nt = Nout >> 6;                   // 64-wide column tiles
  const int mi = wave / nt, ni = wave % nt;
  if (mi >= (N >> 5)) return;                 // uniform per wave (32-row tiles)
  const int m0 = mi << 5, n0 = ni << 6;
  const int am    = lane & 15;                // A: lane -> row M
  const int akoff = (lane < 16) ? 0 : 8;      // A 16-bit layout K offsets
  const int bn    = lane & 15;                // B: lane -> col N
  const int bkoff = (lane < 16) ? 0 : 8;      // B 16-bit layout K offsets (same as A)

  const half_t* arow[2];
  const half_t* brow[4];
#pragma unroll
  for (int t = 0; t < 2; ++t)
    arow[t] = X + (size_t)(m0 + t * 16 + am) * Kdim;
#pragma unroll
  for (int t = 0; t < 4; ++t)
    brow[t] = Wt + (size_t)(n0 + t * 16 + bn) * Kdim;

  v8f acc[2][4];
#pragma unroll
  for (int i = 0; i < 2; ++i)
#pragma unroll
    for (int j = 0; j < 4; ++j) acc[i][j] = (v8f){};

  for (int kk = 0; kk < Kdim; kk += 32) {
    v16h a[2], b[4];
#pragma unroll
    for (int t = 0; t < 2; ++t)
      a[t] = load16(arow[t] + kk + akoff, arow[t] + kk + akoff + 16);
#pragma unroll
    for (int t = 0; t < 4; ++t)
      b[t] = load16(brow[t] + kk + bkoff, brow[t] + kk + bkoff + 16);
#pragma unroll
    for (int i = 0; i < 2; ++i)
#pragma unroll
      for (int j = 0; j < 4; ++j)
        acc[i][j] = __builtin_amdgcn_wmma_f32_16x16x32_f16(false, a[i], false, b[j],
                                                           (short)0, acc[i][j], false, false);
  }

  const int mh = (lane >> 4) << 3;            // C/D layout: lanes>=16 hold M=8..15
#pragma unroll
  for (int j = 0; j < 4; ++j) {
    const float bval = bias[n0 + j * 16 + bn];
#pragma unroll
    for (int i = 0; i < 2; ++i) {
#pragma unroll
      for (int v = 0; v < 8; ++v) {
        float val = acc[i][j][v] + bval;
        if (applySin) val = __sinf(val);
        Y[(size_t)(m0 + i * 16 + mh + v) * Nout + (n0 + j * 16 + bn)] = (half_t)val;
      }
    }
  }
}

// ---------------------- fused attention (flash style) ----------------------
// Each wave: one 16-row Q tile (all 16 K-fragments hoisted to registers,
// 128 VGPRs, loop-invariant) x one column split. Per 16-col K chunk:
// 16 streamed B-frags feed 16 WMMAs; sigmoid in f32, fold V (3 cols) into
// per-lane accumulators; shuffle-reduce over columns; atomicAdd into d_out.
__global__ void __launch_bounds__(256, 1)
k_attn(const half_t* __restrict__ Qh, const half_t* __restrict__ Kh,
       const float* __restrict__ V, float* __restrict__ out,
       int chunksPerSplit) {
  const int wave = blockIdx.x * (blockDim.x >> 5) + (threadIdx.x >> 5);
  const int lane = threadIdx.x & 31;
  const int qi = wave / NSPLIT_;              // 16-row Q tile index
  const int cs = wave % NSPLIT_;
  const int m0 = qi << 4;
  const int n     = lane & 15;
  const int akoff = (lane < 16) ? 0 : 8;
  const int bkoff = (lane < 16) ? 0 : 8;

  // Hoist the 16x512 Q fragment set (loop-invariant over all K chunks).
  v16h qf[16];
  {
    const half_t* arow = Qh + (size_t)(m0 + n) * KD_;
#pragma unroll
    for (int k16 = 0; k16 < 16; ++k16)
      qf[k16] = load16(arow + k16 * 32 + akoff, arow + k16 * 32 + akoff + 16);
  }

  float acc0[8], acc1[8], acc2[8];
#pragma unroll
  for (int v = 0; v < 8; ++v) { acc0[v] = 0.f; acc1[v] = 0.f; acc2[v] = 0.f; }

  for (int c = 0; c < chunksPerSplit; ++c) {
    const int colbase = (cs * chunksPerSplit + c) << 4;
    const half_t* brow = Kh + (size_t)(colbase + n) * KD_;
    if (c + 1 < chunksPerSplit)               // gfx1250 global_prefetch_b8
      __builtin_prefetch(brow + (size_t)16 * KD_, 0, 1);

    v8f s = {};
#pragma unroll
    for (int k16 = 0; k16 < 16; ++k16) {
      v16h b = load16(brow + k16 * 32 + bkoff, brow + k16 * 32 + bkoff + 16);
      s = __builtin_amdgcn_wmma_f32_16x16x32_f16(false, qf[k16], false, b,
                                                 (short)0, s, false, false);
    }

    const float v0 = V[(colbase + n) * 3 + 0];
    const float v1 = V[(colbase + n) * 3 + 1];
    const float v2 = V[(colbase + n) * 3 + 2];
#pragma unroll
    for (int v = 0; v < 8; ++v) {
      float sg = 1.0f / (1.0f + __expf(-s[v]));   // sigmoid in f32
      acc0[v] += sg * v0;
      acc1[v] += sg * v1;
      acc2[v] += sg * v2;
    }
  }

  // Reduce over the 16 columns: xor masks 1,2,4,8 stay inside each half-wave.
#pragma unroll
  for (int mask = 1; mask < 16; mask <<= 1) {
#pragma unroll
    for (int v = 0; v < 8; ++v) {
      acc0[v] += __shfl_xor(acc0[v], mask, 32);
      acc1[v] += __shfl_xor(acc1[v], mask, 32);
      acc2[v] += __shfl_xor(acc2[v], mask, 32);
    }
  }

  if ((lane & 15) == 0) {
    const int mh = (lane >> 4) << 3;
#pragma unroll
    for (int v = 0; v < 8; ++v) {
      const int row = m0 + mh + v;
      atomicAdd(&out[row * 3 + 0], acc0[v]);
      atomicAdd(&out[row * 3 + 1], acc1[v]);
      atomicAdd(&out[row * 3 + 2], acc2[v]);
    }
  }
}

// ------------------------------- launcher ----------------------------------

extern "C" void kernel_launch(void* const* d_in, const int* in_sizes, int n_in,
                              void* d_out, int out_size, void* d_ws, size_t ws_size,
                              hipStream_t stream) {
  (void)in_sizes; (void)n_in; (void)ws_size;
  const float* x   = (const float*)d_in[0];
  const float* c   = (const float*)d_in[1];
  const float* QW0 = (const float*)d_in[2];  const float* Qb0 = (const float*)d_in[3];
  const float* QWh = (const float*)d_in[4];  const float* Qbh = (const float*)d_in[5];
  const float* QWo = (const float*)d_in[6];  const float* Qbo = (const float*)d_in[7];
  const float* KW0 = (const float*)d_in[8];  const float* Kb0 = (const float*)d_in[9];
  const float* KWh = (const float*)d_in[10]; const float* Kbh = (const float*)d_in[11];
  const float* KWo = (const float*)d_in[12]; const float* Kbo = (const float*)d_in[13];
  const float* VW0 = (const float*)d_in[14]; const float* Vb0 = (const float*)d_in[15];
  const float* VWh = (const float*)d_in[16]; const float* Vbh = (const float*)d_in[17];
  const float* VWo = (const float*)d_in[18]; const float* Vbo = (const float*)d_in[19];

  // --- workspace carve-out (all 256B aligned) ---
  char* ws = (char*)d_ws;
  size_t off = 0;
  auto carve = [&](size_t bytes) -> void* {
    void* p = ws + off;
    off += (bytes + 255) & ~(size_t)255;
    return p;
  };
  const size_t actB = (size_t)NX_ * HD_ * sizeof(half_t);     // 8 MB
  half_t* bufA = (half_t*)carve(actB);
  half_t* bufB = (half_t*)carve(actB);
  half_t* Qf16 = (half_t*)carve(actB);
  half_t* Kf16 = (half_t*)carve(actB);
  float*  Vf32 = (float*) carve((size_t)NC_ * YD_ * sizeof(float));
  const size_t whB = (size_t)NHID_ * HD_ * HD_ * sizeof(half_t);
  const size_t woB = (size_t)HD_ * HD_ * sizeof(half_t);
  half_t* WhQ = (half_t*)carve(whB);
  half_t* WoQ = (half_t*)carve(woB);
  half_t* WhK = (half_t*)carve(whB);
  half_t* WoK = (half_t*)carve(woB);
  half_t* WhV = (half_t*)carve(whB);

  const int THR = 256;

  // 0) zero output (attention accumulates with atomics)
  k_zero_f32<<<(out_size + THR - 1) / THR, THR, 0, stream>>>((float*)d_out, out_size);

  // 1) convert + transpose weights to f16 [out][in]
  const int wElems = HD_ * HD_;
  const int wBlocks = (wElems + THR - 1) / THR;
  for (int l = 0; l < NHID_; ++l) {
    k_cvt_t<<<wBlocks, THR, 0, stream>>>(QWh + (size_t)l * wElems, WhQ + (size_t)l * wElems, HD_, HD_);
    k_cvt_t<<<wBlocks, THR, 0, stream>>>(KWh + (size_t)l * wElems, WhK + (size_t)l * wElems, HD_, HD_);
    k_cvt_t<<<wBlocks, THR, 0, stream>>>(VWh + (size_t)l * wElems, WhV + (size_t)l * wElems, HD_, HD_);
  }
  k_cvt_t<<<wBlocks, THR, 0, stream>>>(QWo, WoQ, HD_, HD_);
  k_cvt_t<<<wBlocks, THR, 0, stream>>>(KWo, WoK, HD_, HD_);

  // 2) MLP pipelines
  const int l0Blocks = (NX_ * HD_ + THR - 1) / THR;
  const int gemmWaves  = (NX_ / 32) * (HD_ / 64);       // 2048 waves (32x64 tiles)
  const int gemmBlocks = gemmWaves * 32 / THR;          // 256 blocks of 8 waves

  // Q = MLP(x) -> Qf16
  k_layer0<<<l0Blocks, THR, 0, stream>>>(x, QW0, Qb0, bufA, NX_);
  {
    half_t* src = bufA; half_t* dst = bufB;
    for (int l = 0; l < NHID_; ++l) {
      k_gemm_sin<<<gemmBlocks, THR, 0, stream>>>(src, WhQ + (size_t)l * wElems,
                                                 Qbh + l * HD_, dst, NX_, HD_, HD_, 1);
      half_t* t = src; src = dst; dst = t;
    }
    k_gemm_sin<<<gemmBlocks, THR, 0, stream>>>(src, WoQ, Qbo, Qf16, NX_, HD_, HD_, 0);
  }

  // K = MLP(c) -> Kf16
  k_layer0<<<l0Blocks, THR, 0, stream>>>(c, KW0, Kb0, bufA, NC_);
  {
    half_t* src = bufA; half_t* dst = bufB;
    for (int l = 0; l < NHID_; ++l) {
      k_gemm_sin<<<gemmBlocks, THR, 0, stream>>>(src, WhK + (size_t)l * wElems,
                                                 Kbh + l * HD_, dst, NC_, HD_, HD_, 1);
      half_t* t = src; src = dst; dst = t;
    }
    k_gemm_sin<<<gemmBlocks, THR, 0, stream>>>(src, WoK, Kbo, Kf16, NC_, HD_, HD_, 0);
  }

  // V = MLP(c) -> Vf32 (output head is 512->3, VALU)
  k_layer0<<<l0Blocks, THR, 0, stream>>>(c, VW0, Vb0, bufA, NC_);
  {
    half_t* src = bufA; half_t* dst = bufB;
    for (int l = 0; l < NHID_; ++l) {
      k_gemm_sin<<<gemmBlocks, THR, 0, stream>>>(src, WhV + (size_t)l * wElems,
                                                 Vbh + l * HD_, dst, NC_, HD_, HD_, 1);
      half_t* t = src; src = dst; dst = t;
    }
    k_vout<<<(NC_ * YD_ + THR - 1) / THR, THR, 0, stream>>>(src, VWo, Vbo, Vf32, NC_);
  }

  // 3) fused sigmoid-attention: out += sigmoid(Q K^T) V
  const int chunksPerSplit = (NC_ / 16) / NSPLIT_;      // 32
  const int attnWaves  = (NX_ / 16) * NSPLIT_;          // 8192 (16 Q rows/wave)
  const int attnBlocks = attnWaves * 32 / THR;          // 1024
  k_attn<<<attnBlocks, THR, 0, stream>>>(Qf16, Kf16, Vf32, (float*)d_out, chunksPerSplit);
}
